// WrapperModel_80616536146381
// MI455X (gfx1250) — matrix operand order repaired and checked by reading, not compiled, not run on hardware
//
#include <hip/hip_runtime.h>
#include <hip/hip_bf16.h>

// ---------------------------------------------------------------------------
// Problem constants (match reference)
// ---------------------------------------------------------------------------
#define BATCH      1024
#define N_ITEMS    100000
#define N_CLUSTERS 10
#define DIM        64
#define N_COLTILES (N_ITEMS / 16)   // 6250, exact

// Workspace layout:
//   floats:
//     [0,640)            S1[10][64]      cluster-summed W1
//     [640,650)          counts[10]
//     [1024,66560)       h[1024][64]
//     [66560,76800)      sums[1024][10]  raw exp sums per (row,cluster)
//   dwords (after float region):
//     w2pack[6250][32][16]  pre-packed bf16 B operands (12.8 MB)
#define WS_S1     0
#define WS_COUNTS 640
#define WS_H      1024
#define WS_SUMS   (1024 + BATCH * DIM)
#define WS_FLOATS (WS_SUMS + BATCH * N_CLUSTERS)          // 76800 floats
#define W2P_DWORDS ((size_t)N_COLTILES * 32 * 16)         // 3,200,000 dwords

typedef __attribute__((ext_vector_type(16))) __bf16        v16bf;
typedef __attribute__((ext_vector_type(8)))  float         v8f;
typedef __attribute__((ext_vector_type(8)))  unsigned int  v8u;

// Pack two f32 into one dword of bf16 (round-to-nearest-even).
__device__ __forceinline__ unsigned int bf2(float lo, float hi) {
    unsigned int a = __float_as_uint(lo);
    unsigned int b = __float_as_uint(hi);
    a = (a + 0x7FFFu + ((a >> 16) & 1u)) >> 16;
    b = (b + 0x7FFFu + ((b >> 16) & 1u)) >> 16;
    return (a & 0xFFFFu) | (b << 16);
}

// A-matrix (16x32 bf16) operand for one K-half.
// Lanes 0-15 (hi=0): VGPR0-7 hold K=kb+0..15.
// Lanes 16-31 (hi=1): K=kb+16..31 (matches w2pack B layout).
__device__ __forceinline__ v16bf make_a(const float* __restrict__ hrow, int kb, int hi) {
    v8u u;
    const int o = kb + hi * 16;
#pragma unroll
    for (int j = 0; j < 4; ++j) u[j]     = bf2(hrow[o + 2*j],      hrow[o + 2*j + 1]);
#pragma unroll
    for (int j = 0; j < 4; ++j) u[4 + j] = bf2(hrow[o + 8 + 2*j], hrow[o + 8 + 2*j + 1]);
    return __builtin_bit_cast(v16bf, u);
}

// ---------------------------------------------------------------------------
// Kernel 0: zero the accumulated workspace regions
// ---------------------------------------------------------------------------
__global__ void k_zero(float* ws, int n) {
    int i = blockIdx.x * 256 + threadIdx.x;
    if (i < n) ws[i] = 0.0f;
}

// ---------------------------------------------------------------------------
// Kernel 1: S1[c][d] = sum_{i: cluster[i]==c} W1[i][d];  counts[c] = |{i}|
// Block = 256 threads = 4 item-groups x 64 columns, 256 items per block.
// ---------------------------------------------------------------------------
__global__ void k_cluster_reduce(const float* __restrict__ W1,
                                 const int* __restrict__ cluster,
                                 float* __restrict__ S1,
                                 float* __restrict__ counts) {
    __shared__ float ls[N_CLUSTERS * DIM];
    __shared__ float lc[N_CLUSTERS];
    const int tid = threadIdx.x;
    for (int i = tid; i < N_CLUSTERS * DIM; i += 256) ls[i] = 0.0f;
    if (tid < N_CLUSTERS) lc[tid] = 0.0f;
    __syncthreads();

    const int d    = tid & 63;
    const int g    = tid >> 6;
    const int base = blockIdx.x * 256;
#pragma unroll 4
    for (int ii = g; ii < 256; ii += 4) {
        const int i = base + ii;
        if (i < N_ITEMS) {
            const int c = cluster[i];
            atomicAdd(&ls[c * DIM + d], W1[(size_t)i * DIM + d]);
            if (d == 0) atomicAdd(&lc[c], 1.0f);
        }
    }
    __syncthreads();
    for (int i = tid; i < N_CLUSTERS * DIM; i += 256) atomicAdd(&S1[i], ls[i]);
    if (tid < N_CLUSTERS) atomicAdd(&counts[tid], lc[tid]);
}

// ---------------------------------------------------------------------------
// Kernel 2: h[b][d] = sum_c input[b][c] * S1[c][d]   (1024x10 @ 10x64)
// ---------------------------------------------------------------------------
__global__ void k_small_gemm(const float* __restrict__ in,
                             const float* __restrict__ S1,
                             float* __restrict__ h) {
    const int d = threadIdx.x & 63;
    const int b = blockIdx.x * 4 + (threadIdx.x >> 6);
    float acc = 0.0f;
#pragma unroll
    for (int c = 0; c < N_CLUSTERS; ++c)
        acc = fmaf(in[b * N_CLUSTERS + c], S1[c * DIM + d], acc);
    h[b * DIM + d] = acc;
}

// ---------------------------------------------------------------------------
// Kernel 2.5: pre-pack W2 (f32, 64 x 100000 row-major) into the exact
// per-lane WMMA B-operand register image, bf16:
//   w2p[(t*32 + lane)*16 + kk*8 + j] = bf16x2{ W2[K][n], W2[K+1][n] }
// where n = t*16 + (lane&15), hi = lane>>4, K = kk*32 + hi*16 + 2j.
// Thread = (K-pair, column): W2 reads fully coalesced; one b32 store.
// ---------------------------------------------------------------------------
__global__ void k_pack_w2(const float* __restrict__ W2,
                          unsigned int* __restrict__ w2p) {
    const int n = blockIdx.x * 256 + threadIdx.x;   // column 0..99999
    if (n >= N_ITEMS) return;
    const int K  = blockIdx.y * 2;                  // even K, 0..62
    const unsigned int u = bf2(W2[(size_t)K       * N_ITEMS + n],
                               W2[(size_t)(K + 1) * N_ITEMS + n]);
    const int kk   = K >> 5;                        // which WMMA (K half: 0..31 / 32..63)
    const int k2   = K & 31;
    const int hi   = (k2 >> 4) & 1;                 // lane half
    const int j    = (k2 & 15) >> 1;                // dword within group of 8
    const int t    = n >> 4;                        // column tile
    const int lane = hi * 16 + (n & 15);
    w2p[(size_t)(t * 32 + lane) * 16 + kk * 8 + j] = u;
}

// ---------------------------------------------------------------------------
// Kernel 3: fused  exp(h @ W2)  segment-summed by cluster.
// Block = 256 threads (8 waves). blockIdx.y = 16-row tile of h.
// A tile built once per wave; B operands are direct b128 loads from w2pack.
// 2x v_wmma_f32_16x16x32_bf16 per 16-column tile (K=64), then per-lane
// exp + LDS float atomics into a [16][10] partial table.
// ---------------------------------------------------------------------------
#define ITER 8   // column tiles per wave per block

__global__ void k_gemm_softmax(const float* __restrict__ h,
                               const unsigned int* __restrict__ w2p,
                               const int* __restrict__ cluster,
                               float* __restrict__ sums) {
    __shared__ float ls[16 * N_CLUSTERS];
    const int tid  = threadIdx.x;
    const int lane = tid & 31;
    const int wave = tid >> 5;
    if (tid < 16 * N_CLUSTERS) ls[tid] = 0.0f;
    __syncthreads();

    const int rowtile = blockIdx.y;
    const int r  = lane & 15;
    const int hi = lane >> 4;

    // A tile: rows rowtile*16 .. +15, K = 0..63, as two 16x32 bf16 operands.
    const float* hrow = h + (size_t)(rowtile * 16 + r) * DIM;
    const v16bf a0 = make_a(hrow, 0, hi);
    const v16bf a1 = make_a(hrow, 32, hi);

    for (int it = 0; it < ITER; ++it) {
        const int t = (blockIdx.x * ITER + it) * 8 + wave;   // wave-uniform
        if (t >= N_COLTILES) continue;                       // EXEC stays full
        const int n = t * 16 + r;                            // this lane's column

        // B operands: direct 64-byte aligned vector loads (4x b128), no VALU.
        const v8u* bp = (const v8u*)(w2p + (size_t)(t * 32 + lane) * 16);
        const v16bf b0 = __builtin_bit_cast(v16bf, bp[0]);
        const v16bf b1 = __builtin_bit_cast(v16bf, bp[1]);

        v8f acc = {};
        acc = __builtin_amdgcn_wmma_f32_16x16x32_bf16(false, a0, false, b0,
                                                      (short)0, acc, false, false);
        acc = __builtin_amdgcn_wmma_f32_16x16x32_bf16(false, a1, false, b1,
                                                      (short)0, acc, false, false);

        const int cl = cluster[n];
        // C/D layout: lane owns column n; VGPR g = row g (lanes 0-15) or 8+g.
#pragma unroll
        for (int g = 0; g < 8; ++g) {
            const float e = __expf(acc[g]);
            atomicAdd(&ls[(g + 8 * hi) * N_CLUSTERS + cl], e);
        }
    }
    __syncthreads();
    if (tid < 16 * N_CLUSTERS)
        atomicAdd(&sums[rowtile * 16 * N_CLUSTERS + tid], ls[tid]);
}

// ---------------------------------------------------------------------------
// Kernel 4: out[b][c] = s[b][c] / (sum_c' s[b][c']) / max(counts[c],1)
// (softmax denominator == sum of all cluster sums)
// ---------------------------------------------------------------------------
__global__ void k_final(const float* __restrict__ sums,
                        const float* __restrict__ counts,
                        float* __restrict__ out) {
    const int b = blockIdx.x * 256 + threadIdx.x;
    if (b >= BATCH) return;
    float s[N_CLUSTERS];
    float tot = 0.0f;
#pragma unroll
    for (int c = 0; c < N_CLUSTERS; ++c) { s[c] = sums[b * N_CLUSTERS + c]; tot += s[c]; }
    const float inv = 1.0f / tot;
#pragma unroll
    for (int c = 0; c < N_CLUSTERS; ++c)
        out[b * N_CLUSTERS + c] = s[c] * inv / fmaxf(counts[c], 1.0f);
}

// ---------------------------------------------------------------------------
extern "C" void kernel_launch(void* const* d_in, const int* in_sizes, int n_in,
                              void* d_out, int out_size, void* d_ws, size_t ws_size,
                              hipStream_t stream) {
    const float* input   = (const float*)d_in[0];   // (1024, 10)
    const int*   cluster = (const int*)  d_in[1];   // (100000,)
    const float* W1      = (const float*)d_in[2];   // (100000, 64)
    const float* W2      = (const float*)d_in[3];   // (64, 100000)
    float*       out     = (float*)d_out;           // (1024, 10)
    float*       ws      = (float*)d_ws;

    float*        S1     = ws + WS_S1;
    float*        counts = ws + WS_COUNTS;
    float*        hbuf   = ws + WS_H;
    float*        sums   = ws + WS_SUMS;
    unsigned int* w2p    = (unsigned int*)(ws + WS_FLOATS);  // 12.8 MB packed B

    k_zero<<<(WS_FLOATS + 255) / 256, 256, 0, stream>>>(ws, WS_FLOATS);

    k_cluster_reduce<<<(N_ITEMS + 255) / 256, 256, 0, stream>>>(W1, cluster, S1, counts);

    k_small_gemm<<<BATCH / 4, 256, 0, stream>>>(input, S1, hbuf);

    dim3 gridp((N_ITEMS + 255) / 256, DIM / 2);     // (391, 32)
    k_pack_w2<<<gridp, 256, 0, stream>>>(W2, w2p);

    dim3 grid3((N_COLTILES + 8 * ITER - 1) / (8 * ITER), BATCH / 16);  // (98, 64)
    k_gemm_softmax<<<grid3, 256, 0, stream>>>(hbuf, w2p, cluster, sums);

    k_final<<<(BATCH + 255) / 256, 256, 0, stream>>>(sums, counts, out);
}
